// AdvancedCRSN_v2_1477468750582
// MI455X (gfx1250) — compile-verified, matrix operand order checked by reading, not run on hardware
//
#include <hip/hip_runtime.h>

#define EPSF 1e-6f
#define TAUF 0.2f
#define NSYM 64
#define MAXD 8
#define ACTT 0.9999f
#define BATCHN 8192
#define DIMN 512
#define DIM2 1024
#define VOCABN 32000

typedef __attribute__((ext_vector_type(16))) __bf16 v16bf;
typedef __attribute__((ext_vector_type(8)))  __bf16 v8bf;
typedef __attribute__((ext_vector_type(8)))  float  v8f;

// ---- WMMA helpers -------------------------------------------------------

__device__ inline v8f wmma_bf16(v16bf a, v16bf b, v8f c) {
  // D = A(16x32) * B(32x16) + C, f32 accumulate
  return __builtin_amdgcn_wmma_f32_16x16x32_bf16(
      false, a, false, b, (short)0, c, false, false);
}

// Load a per-lane v16bf A/B fragment for the 16-bit 16x32 layout:
// elements 0..7  = K = kh+0..kh+7   (contiguous)
// elements 8..15 = K = kh+16..kh+23 (contiguous)
__device__ inline v16bf ldtile(const __bf16* p) {
  v8bf lo = *reinterpret_cast<const v8bf*>(p);
  v8bf hi = *reinterpret_cast<const v8bf*>(p + 16);
  v16bf r;
#pragma unroll
  for (int i = 0; i < 8; ++i) { r[i] = lo[i]; r[i + 8] = hi[i]; }
  return r;
}

__device__ inline void async_b128(unsigned lds, const __bf16* g) {
  asm volatile("global_load_async_to_lds_b128 %0, %1, off"
               :: "v"(lds), "v"((unsigned long long)(size_t)g)
               : "memory");
}

// ---- small utility kernels ----------------------------------------------

__global__ void k_cvt_bf16(const float* __restrict__ src,
                           __bf16* __restrict__ dst, int n) {
  int i = blockIdx.x * blockDim.x + threadIdx.x;
  int stride = gridDim.x * blockDim.x;
  for (; i < n; i += stride) dst[i] = (__bf16)src[i];
}

__global__ void k_fill(float* __restrict__ p, float v, int n) {
  int i = blockIdx.x * blockDim.x + threadIdx.x;
  int stride = gridDim.x * blockDim.x;
  for (; i < n; i += stride) p[i] = v;
}

__global__ __launch_bounds__(256) void k_cbsq(const float* __restrict__ cb,
                                              float* __restrict__ cbsq) {
  __shared__ float red[256];
  int s = blockIdx.x, t = threadIdx.x;
  float a = 0.f;
  for (int k = t; k < DIM2; k += 256) {
    float v = cb[(size_t)s * DIM2 + k];
    a += v * v;
  }
  red[t] = a; __syncthreads();
  for (int w = 128; w > 0; w >>= 1) {
    if (t < w) red[t] += red[t + w];
    __syncthreads();
  }
  if (t == 0) cbsq[s] = red[0];
}

// ---- embedding ----------------------------------------------------------

__global__ __launch_bounds__(256) void k_embed(
    const int* __restrict__ ids, const float* __restrict__ emag,
    const float* __restrict__ eph, float* __restrict__ zr,
    float* __restrict__ zi, __bf16* __restrict__ zrb,
    __bf16* __restrict__ zib) {
  __shared__ float red[256];
  __shared__ float s_norm;
  int b = blockIdx.x, t = threadIdx.x;
  int idx = ids[b];
  const float* mg = emag + (size_t)idx * DIMN;
  const float* ph = eph + (size_t)idx * DIMN;
  float r0 = mg[t], r1 = mg[t + 256];
  red[t] = r0 * r0 + r1 * r1; __syncthreads();
  for (int w = 128; w > 0; w >>= 1) {
    if (t < w) red[t] += red[t + w];
    __syncthreads();
  }
  if (t == 0) s_norm = sqrtf(red[0]);
  __syncthreads();
  float inv = 1.f / (s_norm + EPSF);
  float p0 = ph[t], p1 = ph[t + 256];
  float a0 = r0 * inv, a1 = r1 * inv;
  float zr0 = a0 * cosf(p0), zi0 = a0 * sinf(p0);
  float zr1 = a1 * cosf(p1), zi1 = a1 * sinf(p1);
  size_t o = (size_t)b * DIMN;
  zr[o + t] = zr0; zr[o + t + 256] = zr1;
  zi[o + t] = zi0; zi[o + t + 256] = zi1;
  zrb[o + t] = (__bf16)zr0; zrb[o + t + 256] = (__bf16)zr1;
  zib[o + t] = (__bf16)zi0; zib[o + t + 256] = (__bf16)zi1;
}

// ---- complex GEMM:  ar = Zr Wr^T - Zi Wi^T ; ai = Zi Wr^T + Zr Wi^T -----
// M=8192, N=512, K=512. Block = 8 waves sharing one 16-row A strip (zr+zi)
// staged in LDS via async copies (double buffered, K=64 per stage).
// Wave w computes N-tile (blk%4)*8 + w.

__global__ __launch_bounds__(256) void k_cgemm(
    const __bf16* __restrict__ zrb, const __bf16* __restrict__ zib,
    const __bf16* __restrict__ wrb, const __bf16* __restrict__ wib,
    float* __restrict__ ar, float* __restrict__ ai) {
  __shared__ __bf16 stage[2][2048];       // per buf: zr 16x64 | zi 16x64
  int wave = threadIdx.x >> 5, lane = threadIdx.x & 31, t = threadIdx.x;
  int tm = blockIdx.x >> 2;               // 512 M-tiles
  int tn = ((blockIdx.x & 3) << 3) + wave;// 32 N-tiles
  int lr = lane & 15;
  int kh = (lane < 16) ? 0 : 8;
  const __bf16* pwr = wrb + (size_t)(tn * 16 + lr) * DIMN + kh;
  const __bf16* pwi = wib + (size_t)(tn * 16 + lr) * DIMN + kh;
  unsigned sbase = (unsigned)(size_t)(&stage[0][0]);

  // async stage issue: 256 threads x 16B move 16x64 zr + 16x64 zi (4KB)
  int im = t >> 7;                        // 0 = zr, 1 = zi
  int irow = (t >> 3) & 15;
  int ich = t & 7;                        // 8 x 16B per 128B row
  const __bf16* gsrc0 =
      (im ? zib : zrb) + (size_t)(tm * 16 + irow) * DIMN + ich * 8;
  unsigned ldst = (unsigned)(im * 2048 + irow * 128 + ich * 16);

  v8f rr = {0.f,0.f,0.f,0.f,0.f,0.f,0.f,0.f};
  v8f ii = rr, ir = rr, ri = rr;

  async_b128(sbase + ldst, gsrc0);        // prefetch buffer 0 (k0 = 0)

  for (int k0 = 0; k0 < DIMN; k0 += 64) {
    int cur = (k0 >> 6) & 1;
    bool more = (k0 + 64) < DIMN;
    if (more) {
      async_b128(sbase + (unsigned)((cur ^ 1) * 4096) + ldst, gsrc0 + k0 + 64);
      asm volatile("s_wait_asynccnt 0x1" ::: "memory");
    } else {
      asm volatile("s_wait_asynccnt 0x0" ::: "memory");
    }
    __syncthreads();

    const __bf16* spR = &stage[cur][0];
    const __bf16* spI = &stage[cur][1024];
#pragma unroll
    for (int ks = 0; ks < 64; ks += 32) {
      v16bf Br = ldtile(pwr + k0 + ks);
      v16bf Bi = ldtile(pwi + k0 + ks);
      v16bf Ar = ldtile(spR + lr * 64 + ks + kh);
      v16bf Ai = ldtile(spI + lr * 64 + ks + kh);
      rr = wmma_bf16(Ar, Br, rr);
      ii = wmma_bf16(Ai, Bi, ii);
      ir = wmma_bf16(Ai, Br, ir);
      ri = wmma_bf16(Ar, Bi, ri);
    }
    __syncthreads();
  }
  int n = tn * 16 + lr;
  int mb = tm * 16 + ((lane < 16) ? 0 : 8);
#pragma unroll
  for (int v = 0; v < 8; ++v) {
    ar[(size_t)(mb + v) * DIMN + n] = rr[v] - ii[v];
    ai[(size_t)(mb + v) * DIMN + n] = ir[v] + ri[v];
  }
}

// ---- per-row pointwise step ---------------------------------------------

__device__ inline float blk_sum(float v, float* red) {
  int t = threadIdx.x;
  red[t] = v; __syncthreads();
  for (int w = 128; w > 0; w >>= 1) {
    if (t < w) red[t] += red[t + w];
    __syncthreads();
  }
  float r = red[0]; __syncthreads();
  return r;
}

__global__ __launch_bounds__(256) void k_step(
    const float* __restrict__ ar_g, const float* __restrict__ ai_g,
    const float* __restrict__ ln_scale, const float* __restrict__ ln_shift,
    const float* __restrict__ mr_bias, const float* __restrict__ halt_w,
    const float* __restrict__ halt_b, const float* __restrict__ codebook,
    const float* __restrict__ cb_sq, const float* __restrict__ adjacency,
    float* __restrict__ zr, float* __restrict__ zi,
    __bf16* __restrict__ zrb, __bf16* __restrict__ zib,
    float* __restrict__ halting, float* __restrict__ remain,
    int* __restrict__ sym, float* __restrict__ zw_r, float* __restrict__ zw_i,
    float* __restrict__ o_sym, float* __restrict__ o_ponder,
    float* __restrict__ o_vq, __bf16* __restrict__ featb, int step) {
  __shared__ float sh_cr[DIMN], sh_ci[DIMN], red[256];
  __shared__ float sh_d2[NSYM], sh_prob[NSYM];
  __shared__ float s_p;
  __shared__ int s_min;
  int b = blockIdx.x, t = threadIdx.x;
  const float* arp = ar_g + (size_t)b * DIMN;
  const float* aip = ai_g + (size_t)b * DIMN;
  int j[2] = { t, t + 256 };
  float arv[2], aiv[2], magv[2], crv[2], civ[2];
  float p1 = 0.f, p2 = 0.f;
#pragma unroll
  for (int u = 0; u < 2; ++u) {
    float x = arp[j[u]], y = aip[j[u]];
    arv[u] = x; aiv[u] = y;
    float h = sqrtf(x * x + y * y);
    float m = h + EPSF;
    magv[u] = m;
    p1 += m; p2 += m * m;
  }
  float sum = blk_sum(p1, red);
  float ssq = blk_sum(p2, red);
  float mean = sum * (1.f / 512.f);
  float var = (ssq - 512.f * mean * mean) * (1.f / 511.f);
  float isd = 1.f / sqrtf(var + EPSF);
  float hp = 0.f, zp = 0.f;
#pragma unroll
  for (int u = 0; u < 2; ++u) {
    int jj = j[u];
    float nm = (magv[u] - mean) * isd * ln_scale[jj] + ln_shift[jj];
    float h = magv[u] - EPSF;
    float c, s;
    if (h > 0.f) { c = arv[u] / h; s = aiv[u] / h; } else { c = 1.f; s = 0.f; }
    float nr = nm * c, ni = nm * s;
    float norm = sqrtf(nr * nr + ni * ni) + EPSF;
    float sf = fmaxf(norm + mr_bias[jj], 0.f) / norm;
    crv[u] = nr * sf; civ[u] = ni * sf;
    sh_cr[jj] = crv[u]; sh_ci[jj] = civ[u];
    hp += crv[u] * halt_w[jj] + civ[u] * halt_w[DIMN + jj];
    zp += crv[u] * crv[u] + civ[u] * civ[u];
  }
  float hdot = blk_sum(hp, red);   // also syncs sh_cr/sh_ci
  float z2 = blk_sum(zp, red);
  float p_halt = 1.f / (1.f + expf(-(hdot + halt_b[0])));
  int prev = sym[b];

  // codebook dot products: 4 threads per symbol, 256 dims each
  {
    int sid = t >> 2, prt = t & 3;
    const float* cb = codebook + (size_t)sid * DIM2;
    float dp = 0.f;
    int j0 = prt * 128;
    for (int q = 0; q < 128; ++q) {
      int jj = j0 + q;
      dp += sh_cr[jj] * cb[jj] + sh_ci[jj] * cb[DIMN + jj];
    }
    red[t] = dp; __syncthreads();
    if (t < NSYM) {
      float dot = red[4 * t] + red[4 * t + 1] + red[4 * t + 2] + red[4 * t + 3];
      sh_d2[t] = z2 + cb_sq[t] - 2.f * dot;
    }
    __syncthreads();
  }

  if (t == 0) {
    float best = 3.4e38f; int bi = 0;
    for (int s = 0; s < NSYM; ++s) {
      float dv = sh_d2[s];
      if (step > 0) {
        float a = adjacency[(size_t)prev * NSYM + s];
        dv -= 0.5f * (1.f / (1.f + expf(-a)));
      }
      if (dv < best) { best = dv; bi = s; }
    }
    s_min = bi;
    float mx = -3.4e38f;
    for (int s = 0; s < NSYM; ++s) {
      float d = -sqrtf(fmaxf(sh_d2[s], 0.f)) * (1.f / TAUF);
      sh_prob[s] = d;
      mx = fmaxf(mx, d);
    }
    float se = 0.f;
    for (int s = 0; s < NSYM; ++s) {
      float e = expf(sh_prob[s] - mx);
      sh_prob[s] = e; se += e;
    }
    float inv = 1.f / se;
    for (int s = 0; s < NSYM; ++s) sh_prob[s] *= inv;
  }
  __syncthreads();

  int mi = s_min;
  const float* cbq = codebook + (size_t)mi * DIM2;
  float zq[4];
  float vqp = 0.f;
#pragma unroll
  for (int u = 0; u < 4; ++u) {
    int jj = t + u * 256;
    float acc = 0.f;
    for (int s = 0; s < NSYM; ++s)
      acc += sh_prob[s] * codebook[(size_t)s * DIM2 + jj];
    zq[u] = acc;
    float zf = (jj < DIMN) ? sh_cr[jj] : sh_ci[jj - DIMN];
    float dq = cbq[jj] - zf;
    vqp += dq * dq;
  }
  float vqsum = blk_sum(vqp, red);

  float znr[2], zni[2];
  znr[0] = 0.7f * crv[0] + 0.3f * zq[0];
  znr[1] = 0.7f * crv[1] + 0.3f * zq[1];
  zni[0] = 0.7f * civ[0] + 0.3f * zq[2];
  zni[1] = 0.7f * civ[1] + 0.3f * zq[3];

  if (t == 0) {
    float h = halting[b], rem = remain[b];
    float still = (h < ACTT) ? 1.f : 0.f;
    float p = fminf(fmaxf(p_halt * still, 1e-6f), 1.f);
    if (step == MAXD - 1) p = rem;
    halting[b] = h + p;
    remain[b] = rem - p;
    s_p = p;
    sym[b] = mi;
    o_sym[b] = (float)mi;
    atomicAdd(o_ponder, still * (1.f / (float)BATCHN));
    atomicAdd(o_vq, 1.25f * vqsum * (1.f / ((float)BATCHN * (float)DIM2)));
  }
  __syncthreads();
  float p = s_p;
  size_t o = (size_t)b * DIMN;
#pragma unroll
  for (int u = 0; u < 2; ++u) {
    int jj = j[u];
    float wr2 = zw_r[o + jj] + p * znr[u];
    float wi2 = zw_i[o + jj] + p * zni[u];
    zw_r[o + jj] = wr2; zw_i[o + jj] = wi2;
    zr[o + jj] = znr[u]; zi[o + jj] = zni[u];
    zrb[o + jj] = (__bf16)znr[u]; zib[o + jj] = (__bf16)zni[u];
    if (step == MAXD - 1) {
      featb[(size_t)b * DIM2 + jj] = (__bf16)wr2;
      featb[(size_t)b * DIM2 + DIMN + jj] = (__bf16)wi2;
    }
  }
}

// ---- decoder GEMM: logits = feat(8192x1024) @ dec_w^T(1024x32000) + b ---
// Block: 8 waves share one 64-row A strip staged in LDS via async copies
// (double buffered, K=64 per stage => 8 WMMAs per barrier pair).
// Each wave computes a 64M x 16N strip; B tiles are per-wave global loads.

__global__ __launch_bounds__(256) void k_dec(
    const __bf16* __restrict__ featb, const __bf16* __restrict__ decwb,
    const float* __restrict__ dec_b, float* __restrict__ logits) {
  __shared__ __bf16 stageA[2][64 * 64];   // 2 x 8KB double buffer
  int wave = threadIdx.x >> 5, lane = threadIdx.x & 31, t = threadIdx.x;
  int sm = blockIdx.x / 250;              // 128 super-rows of 64
  int ng = blockIdx.x % 250;              // 250 groups of 8 N-tiles
  int tn = ng * 8 + wave;                 // this wave's N-tile
  int lr = lane & 15;
  int kh = (lane < 16) ? 0 : 8;
  const __bf16* bp = decwb + (size_t)(tn * 16 + lr) * DIM2 + kh;
  unsigned sbase = (unsigned)(size_t)(&stageA[0][0]);

  // async stage issue: 64 rows x 64 K (8KB) = 256 threads x 2 x 16B
  int irow = t >> 2;                      // 0..63
  int ich = t & 3;                        // chunks ich and ich+4 of 8
  const __bf16* gsrc0 =
      featb + (size_t)(sm * 64 + irow) * DIM2 + ich * 8;
  unsigned ldst = (unsigned)(irow * 128 + ich * 16);

  v8f a0 = {0.f,0.f,0.f,0.f,0.f,0.f,0.f,0.f};
  v8f a1 = a0, a2 = a0, a3 = a0;

  async_b128(sbase + ldst, gsrc0);        // prefetch buffer 0
  async_b128(sbase + ldst + 64, gsrc0 + 32);

  for (int k0 = 0; k0 < DIM2; k0 += 64) {
    int cur = (k0 >> 6) & 1;
    bool more = (k0 + 64) < DIM2;
    if (more) {
      unsigned nb = sbase + (unsigned)((cur ^ 1) * 8192) + ldst;
      async_b128(nb, gsrc0 + k0 + 64);
      async_b128(nb + 64, gsrc0 + k0 + 96);
      asm volatile("s_wait_asynccnt 0x2" ::: "memory");
    } else {
      asm volatile("s_wait_asynccnt 0x0" ::: "memory");
    }
    __syncthreads();                      // stage[cur] visible to all waves

    const __bf16* sp = &stageA[cur][0];
#pragma unroll
    for (int ks = 0; ks < 64; ks += 32) {
      v16bf B = ldtile(bp + k0 + ks);
      v16bf A0 = ldtile(sp + (0 * 16 + lr) * 64 + ks + kh);
      v16bf A1 = ldtile(sp + (1 * 16 + lr) * 64 + ks + kh);
      v16bf A2 = ldtile(sp + (2 * 16 + lr) * 64 + ks + kh);
      v16bf A3 = ldtile(sp + (3 * 16 + lr) * 64 + ks + kh);
      a0 = wmma_bf16(A0, B, a0);
      a1 = wmma_bf16(A1, B, a1);
      a2 = wmma_bf16(A2, B, a2);
      a3 = wmma_bf16(A3, B, a3);
    }
    __syncthreads();                      // reads done before overwrite
  }

  int n = tn * 16 + lr;
  float bias = dec_b[n];
  size_t mrow = (size_t)(sm * 64 + ((lane < 16) ? 0 : 8));
#pragma unroll
  for (int v = 0; v < 8; ++v) {
    logits[(mrow + v) * VOCABN + n]      = a0[v] + bias;
    logits[(mrow + 16 + v) * VOCABN + n] = a1[v] + bias;
    logits[(mrow + 32 + v) * VOCABN + n] = a2[v] + bias;
    logits[(mrow + 48 + v) * VOCABN + n] = a3[v] + bias;
  }
}

// ---- host ---------------------------------------------------------------

extern "C" void kernel_launch(void* const* d_in, const int* in_sizes, int n_in,
                              void* d_out, int out_size, void* d_ws,
                              size_t ws_size, hipStream_t stream) {
  (void)in_sizes; (void)n_in; (void)out_size; (void)ws_size;
  const int*   input_ids = (const int*)d_in[0];
  const float* emb_mag   = (const float*)d_in[1];
  const float* emb_phase = (const float*)d_in[2];
  const float* w_real    = (const float*)d_in[3];
  const float* w_imag    = (const float*)d_in[4];
  const float* ln_scale  = (const float*)d_in[5];
  const float* ln_shift  = (const float*)d_in[6];
  const float* mr_bias   = (const float*)d_in[7];
  const float* halt_w    = (const float*)d_in[8];
  const float* halt_b    = (const float*)d_in[9];
  const float* codebook  = (const float*)d_in[10];
  const float* adjacency = (const float*)d_in[11];
  const float* dec_w     = (const float*)d_in[12];
  const float* dec_b     = (const float*)d_in[13];

  // d_out layout: logits | zw_r | zw_i | sym | ponder | vq
  float* logits = (float*)d_out;
  float* o_zwr  = logits + (size_t)BATCHN * VOCABN;
  float* o_zwi  = o_zwr + (size_t)BATCHN * DIMN;
  float* o_sym  = o_zwi + (size_t)BATCHN * DIMN;
  float* o_pond = o_sym + BATCHN;
  float* o_vq   = o_pond + 1;

  // workspace layout (256B aligned)
  char* ws = (char*)d_ws;
  size_t off = 0;
  auto take = [&](size_t bytes) {
    char* p = ws + off;
    off = (off + bytes + 255) & ~(size_t)255;
    return p;
  };
  float*  zr_f   = (float*)take((size_t)BATCHN * DIMN * 4);
  float*  zi_f   = (float*)take((size_t)BATCHN * DIMN * 4);
  float*  ar_f   = (float*)take((size_t)BATCHN * DIMN * 4);
  float*  ai_f   = (float*)take((size_t)BATCHN * DIMN * 4);
  float*  halting = (float*)take((size_t)BATCHN * 4);
  float*  remain  = (float*)take((size_t)BATCHN * 4);
  int*    sym_ws  = (int*)take((size_t)BATCHN * 4);
  float*  cbsq    = (float*)take(NSYM * 4);
  __bf16* zr_b   = (__bf16*)take((size_t)BATCHN * DIMN * 2);
  __bf16* zi_b   = (__bf16*)take((size_t)BATCHN * DIMN * 2);
  __bf16* wr_b   = (__bf16*)take((size_t)DIMN * DIMN * 2);
  __bf16* wi_b   = (__bf16*)take((size_t)DIMN * DIMN * 2);
  __bf16* feat_b = (__bf16*)take((size_t)BATCHN * DIM2 * 2);
  __bf16* decw_b = (__bf16*)take((size_t)VOCABN * DIM2 * 2);

  // init accumulators
  hipMemsetAsync(o_zwr, 0, (size_t)2 * BATCHN * DIMN * 4, stream);
  hipMemsetAsync(o_pond, 0, 2 * 4, stream);
  hipMemsetAsync(halting, 0, (size_t)BATCHN * 4, stream);
  hipMemsetAsync(sym_ws, 0, (size_t)BATCHN * 4, stream);
  k_fill<<<32, 256, 0, stream>>>(remain, 1.0f, BATCHN);

  // weight conversions + codebook squared norms
  k_cvt_bf16<<<1024, 256, 0, stream>>>(w_real, wr_b, DIMN * DIMN);
  k_cvt_bf16<<<1024, 256, 0, stream>>>(w_imag, wi_b, DIMN * DIMN);
  k_cvt_bf16<<<8192, 256, 0, stream>>>(dec_w, decw_b, VOCABN * DIM2);
  k_cbsq<<<NSYM, 256, 0, stream>>>(codebook, cbsq);

  // embedding
  k_embed<<<BATCHN, 256, 0, stream>>>(input_ids, emb_mag, emb_phase, zr_f,
                                      zi_f, zr_b, zi_b);

  // recurrent steps
  for (int step = 0; step < MAXD; ++step) {
    k_cgemm<<<2048, 256, 0, stream>>>(zr_b, zi_b, wr_b, wi_b, ar_f, ai_f);
    k_step<<<BATCHN, 256, 0, stream>>>(
        ar_f, ai_f, ln_scale, ln_shift, mr_bias, halt_w, halt_b, codebook,
        cbsq, adjacency, zr_f, zi_f, zr_b, zi_b, halting, remain, sym_ws,
        o_zwr, o_zwi, o_sym, o_pond, o_vq, feat_b, step);
  }

  // decoder (async-LDS staged)
  k_dec<<<32000, 256, 0, stream>>>(feat_b, decw_b, dec_b, logits);
}
